// BiMambaBlock_8426725834778
// MI455X (gfx1250) — matrix operand order repaired and checked by reading, not compiled, not run on hardware
//
#include <hip/hip_runtime.h>
#include <stdint.h>

// ---------------------------------------------------------------------------
// BiMamba block for MI455X (gfx1250).
// GEMMs: bf16 operands end-to-end, TDM (tensor_load_to_lds) double-buffered
// staging with LDS padding for aligned ds_load_b128 fragment reads, and
// v_wmma_f32_16x16x32_bf16 compute. Scan: register-resident recurrence.
// ---------------------------------------------------------------------------

#define B_SZ    4
#define LSEQ    2048
#define DMODEL  1024
#define DINNER  2048
#define DSTATE  16
#define DTRANK  64
#define NROWS   (B_SZ * LSEQ)   // 8192 token rows

typedef __attribute__((ext_vector_type(16))) __bf16 v16bf;
typedef __attribute__((ext_vector_type(8)))  __bf16 v8bf;
typedef __attribute__((ext_vector_type(8)))  float  v8f;
typedef __attribute__((ext_vector_type(4)))  unsigned int u32x4;
typedef __attribute__((ext_vector_type(8)))  int    i32x8;
typedef __attribute__((ext_vector_type(4)))  int    i32x4;

#ifndef USE_TDM
#define USE_TDM 1
#endif

__device__ __forceinline__ float silu_f(float v) { return v / (1.0f + __expf(-v)); }

// ---------------------------------------------------------------------------
// Tensor Data Mover: DMA a [rows x cols] tile of `dsz_code`-sized elements
// (row stride = stride_elems) from global into LDS at lds_off, with LDS
// padding pada+1 DWORDs every 2^(padi+1) DWORDs.
// D# packing per CDNA5 ISA.
// ---------------------------------------------------------------------------
__device__ __forceinline__ void tdm_load_tile(const void* gsrc, unsigned lds_off,
                                              int stride_elems, int rows, int cols,
                                              unsigned dsz_code, unsigned padi,
                                              unsigned pada) {
#if USE_TDM
  unsigned long long ga = (unsigned long long)gsrc;
  u32x4 g0;
  g0[0] = 1u;                                      // count=1 (valid user descriptor)
  g0[1] = lds_off;                                 // lds_addr (bytes)
  g0[2] = (unsigned)(ga & 0xffffffffull);          // global_addr[31:0]
  g0[3] = (unsigned)((ga >> 32) & 0x01ffffffull)   // global_addr[56:32]
        | 0x80000000u;                             // type=2 ("image") in [127:126]
  unsigned td0 = (unsigned)cols;                   // tensor_dim0 (elements)
  unsigned td1 = (unsigned)rows;                   // tensor_dim1 (rows)
  i32x8 g1;
  g1[0] = (int)((dsz_code << 16)                   // data_size
              | (1u << 20)                         // pad_enable
              | (padi << 22)                       // pad_interval
              | (pada << 25));                     // pad_amount
  g1[1] = (int)((td0 & 0xffffu) << 16);            // tensor_dim0[15:0]
  g1[2] = (int)(((td0 >> 16) & 0xffffu) | ((td1 & 0xffffu) << 16));
  g1[3] = (int)(((td1 >> 16) & 0xffffu) | (((unsigned)cols & 0xffffu) << 16)); // tile_dim0
  g1[4] = (int)((unsigned)rows & 0xffffu);         // tile_dim1 (tile_dim2 = 0)
  g1[5] = (int)(unsigned)stride_elems;             // tensor_dim0_stride[31:0]
  g1[6] = 0;
  g1[7] = 0;
  i32x4 gz = {0, 0, 0, 0};
#if defined(__clang_major__) && (__clang_major__ >= 23)
  i32x8 gz8 = {0, 0, 0, 0, 0, 0, 0, 0};
  __builtin_amdgcn_tensor_load_to_lds(g0, g1, gz, gz, gz8, 0);
#else
  __builtin_amdgcn_tensor_load_to_lds(g0, g1, gz, gz, 0);
#endif
#else
  (void)gsrc; (void)lds_off; (void)stride_elems; (void)rows; (void)cols;
  (void)dsz_code; (void)padi; (void)pada;
#endif
}

// ---------------------------------------------------------------------------
// WMMA GEMM: C[M,N] (+)= A[M,K] * W[N,K]^T, bf16 in / f32 accumulate+out.
// Block = 256 threads (8 waves), tile BM=64 x BN=64, BK=32.
// Wave w: A-row block mi = w&3, two N blocks (w>>2)*2 + {0,1}.
// TDM double-buffers bf16 tiles into LDS. LDS row stride = 40 bf16 (80 B:
// 16-DWORD rows + 4-DWORD TDM pad) -> 16B-aligned b128 fragment chunks,
// conflict-free (20-bank row skew) across the 16 lanes of a fragment.
// mode: 0 = store, 1 = softplus(acc + bias[n]) store, 2 = accumulate,
//       3 = accumulate with sequence flip (row b*L+l -> b*L+(L-1-l)).
// Cbf (optional): bf16 shadow copy of C (mode 0 only).
// ---------------------------------------------------------------------------
#define BM 64
#define BN 64
#define BK 32
#define LDSTB 40   // LDS row stride in bf16 elements (80 bytes)

__global__ __launch_bounds__(256)
void gemm_wmma_bf16(const __bf16* __restrict__ A, int lda,
                    const __bf16* __restrict__ W, int ldw,
                    float* __restrict__ C, int ldc, __bf16* __restrict__ Cbf,
                    int M, int N, int K,
                    int mode, const float* __restrict__ bias) {
  __shared__ __attribute__((aligned(16))) __bf16 As2[2][BM * LDSTB];
  __shared__ __attribute__((aligned(16))) __bf16 Ws2[2][BN * LDSTB];

  const int tid  = (int)threadIdx.x;
  const int wave = tid >> 5;
  const int lane = tid & 31;
  const int m0 = (int)blockIdx.y * BM;
  const int n0 = (int)blockIdx.x * BN;
  const int mi = wave & 3;
  const int njBase = (wave >> 2) * 2;

  const int tn = (N - n0 > BN) ? BN : (N - n0);
  const int tm = (M - m0 > BM) ? BM : (M - m0);

  v8f acc0 = {}; v8f acc1 = {};

  const int l16 = lane & 15;
  const int hiHalf = (lane >> 4) & 1;       // 0: lanes 0-15, 1: lanes 16-31
  const int kA = hiHalf ? 16 : 0;           // A frag k-base (contiguous k = 16h + s, matches B)
  const int kB = hiHalf ? 16 : 0;           // B frag k-base (contiguous k = 16h + s, matches A)

  int cur = 0;
#if USE_TDM
  if (wave == 0) {  // prologue: stage tile k0=0 into buffer 0
    tdm_load_tile(A + (size_t)m0 * lda, (unsigned)(size_t)&As2[0][0], lda, tm, BK, 1u, 3u, 3u);
    tdm_load_tile(W + (size_t)n0 * ldw, (unsigned)(size_t)&Ws2[0][0], ldw, tn, BK, 1u, 3u, 3u);
  }
#endif

  for (int k0 = 0; k0 < K; k0 += BK) {
#if USE_TDM
    if (wave == 0) {
      if (k0 + BK < K) {   // prefetch next tile into the other buffer
        int kn = k0 + BK;
        tdm_load_tile(A + (size_t)m0 * lda + kn, (unsigned)(size_t)&As2[cur ^ 1][0],
                      lda, tm, BK, 1u, 3u, 3u);
        tdm_load_tile(W + (size_t)n0 * ldw + kn, (unsigned)(size_t)&Ws2[cur ^ 1][0],
                      ldw, tn, BK, 1u, 3u, 3u);
        __builtin_amdgcn_s_wait_tensorcnt(2);  // current tile done; prefetch in flight
      } else {
        __builtin_amdgcn_s_wait_tensorcnt(0);
      }
    }
#else
    for (int i = tid; i < BM * BK; i += 256) {
      int r = i / BK, c = i % BK;
      As2[cur][r * LDSTB + c] = A[(size_t)(m0 + r) * lda + k0 + c];
      Ws2[cur][r * LDSTB + c] = (r < tn) ? W[(size_t)(n0 + r) * ldw + k0 + c] : (__bf16)0.0f;
    }
#endif
    __syncthreads();

    // Fragments via 16B ds_load_b128 chunks (k-contiguous bf16 in LDS).
    const __bf16* ap  = &As2[cur][(mi * 16 + l16) * LDSTB];
    const __bf16* b0p = &Ws2[cur][(njBase * 16 + l16) * LDSTB] + kB;
    const __bf16* b1p = b0p + 16 * LDSTB;
    v8bf alo = *(const v8bf*)(const void*)(ap + kA);
    v8bf ahi = *(const v8bf*)(const void*)(ap + kA + 8);
    v8bf b0lo = *(const v8bf*)(const void*)(b0p);
    v8bf b0hi = *(const v8bf*)(const void*)(b0p + 8);
    v8bf b1lo = *(const v8bf*)(const void*)(b1p);
    v8bf b1hi = *(const v8bf*)(const void*)(b1p + 8);
    v16bf af  = __builtin_shufflevector(alo, ahi, 0,1,2,3,4,5,6,7,8,9,10,11,12,13,14,15);
    v16bf bf0 = __builtin_shufflevector(b0lo, b0hi, 0,1,2,3,4,5,6,7,8,9,10,11,12,13,14,15);
    v16bf bf1 = __builtin_shufflevector(b1lo, b1hi, 0,1,2,3,4,5,6,7,8,9,10,11,12,13,14,15);

    acc0 = __builtin_amdgcn_wmma_f32_16x16x32_bf16(false, af, false, bf0,
                                                   (short)0, acc0, false, false);
    acc1 = __builtin_amdgcn_wmma_f32_16x16x32_bf16(false, af, false, bf1,
                                                   (short)0, acc1, false, false);
    __syncthreads();   // all reads of buf[cur] done before TDM may overwrite it
    cur ^= 1;
  }

  // Epilogue. C/D layout: VGPR r -> m = r + 8*hiHalf, n = lane&15.
  const int mloc = mi * 16 + (hiHalf ? 8 : 0);
#pragma unroll
  for (int r = 0; r < 8; ++r) {
    int gm = m0 + mloc + r;
    if (gm >= M) continue;
    size_t crow = (size_t)gm;
    if (mode == 3) {
      int bb = gm / LSEQ, ll = gm % LSEQ;   // LSEQ is a power of two -> shifts
      crow = (size_t)bb * LSEQ + (LSEQ - 1 - ll);
    }
#pragma unroll
    for (int j = 0; j < 2; ++j) {
      int gn = n0 + (njBase + j) * 16 + l16;
      if (gn >= N) continue;
      float v = j ? acc1[r] : acc0[r];
      float* cp = C + crow * (size_t)ldc + gn;
      if (mode == 0) {
        *cp = v;
        if (Cbf) Cbf[crow * (size_t)ldc + gn] = (__bf16)v;
      } else if (mode == 1) {
        float t = v + bias[gn];
        *cp = (t > 20.0f) ? t : logf(1.0f + __expf(t));
      } else {
        *cp += v;
      }
    }
  }
}

// ---------------------------------------------------------------------------
// Residual init: out = x
// ---------------------------------------------------------------------------
__global__ __launch_bounds__(256)
void copy_kernel(const float* __restrict__ src, float* __restrict__ dst, size_t n) {
  size_t i = (size_t)blockIdx.x * 256 + threadIdx.x;
  if (i < n) dst[i] = src[i];
}

// ---------------------------------------------------------------------------
// f32 -> bf16 cast (weights etc.)
// ---------------------------------------------------------------------------
__global__ __launch_bounds__(256)
void cast_kernel(const float* __restrict__ src, __bf16* __restrict__ dst, size_t n) {
  size_t i = (size_t)blockIdx.x * 256 + threadIdx.x;
  if (i < n) dst[i] = (__bf16)src[i];
}

// ---------------------------------------------------------------------------
// LayerNorm over D=1024; writes bf16 normal-order xn and flipped xnf.
// ---------------------------------------------------------------------------
__global__ __launch_bounds__(256)
void ln_kernel(const float* __restrict__ x, const float* __restrict__ g,
               const float* __restrict__ b, __bf16* __restrict__ xn,
               __bf16* __restrict__ xnf) {
  __shared__ float rs[256];
  __shared__ float rs2[256];
  const int row = (int)blockIdx.x;            // b*L + l
  const int bi = row / LSEQ, l = row % LSEQ;
  const float* xr = x + (size_t)row * DMODEL;
  const int tid = (int)threadIdx.x;

  float s = 0.0f, s2 = 0.0f;
  for (int i = tid; i < DMODEL; i += 256) { float v = xr[i]; s += v; s2 += v * v; }
  rs[tid] = s; rs2[tid] = s2;
  __syncthreads();
  for (int off = 128; off > 0; off >>= 1) {
    if (tid < off) { rs[tid] += rs[tid + off]; rs2[tid] += rs2[tid + off]; }
    __syncthreads();
  }
  const float mu = rs[0] * (1.0f / DMODEL);
  const float var = rs2[0] * (1.0f / DMODEL) - mu * mu;
  const float rstd = rsqrtf(var + 1e-5f);

  __bf16* o1 = xn  + (size_t)row * DMODEL;
  __bf16* o2 = xnf + ((size_t)bi * LSEQ + (LSEQ - 1 - l)) * DMODEL;
  for (int i = tid; i < DMODEL; i += 256) {
    float v = (xr[i] - mu) * rstd * g[i] + b[i];
    __bf16 h = (__bf16)v;
    o1[i] = h;
    o2[i] = h;
  }
}

// ---------------------------------------------------------------------------
// Causal depthwise conv (width 4) + bias + SiLU. xi = xz[:, 0:DINNER]
// (row stride 2*DINNER). Writes f32 (for scan) + bf16 (for x_proj GEMM).
// ---------------------------------------------------------------------------
__global__ __launch_bounds__(256)
void conv_silu_kernel(const float* __restrict__ xz, const float* __restrict__ cw,
                      const float* __restrict__ cb, float* __restrict__ xc,
                      __bf16* __restrict__ xcb) {
  size_t idx = (size_t)blockIdx.x * 256 + threadIdx.x;
  if (idx >= (size_t)NROWS * DINNER) return;
  const int d = (int)(idx % DINNER);
  const size_t rl = idx / DINNER;              // b*L + l
  const int l = (int)(rl % LSEQ);
  const size_t rowBase = rl - l;               // start row of this batch
  float acc = cb[d];
#pragma unroll
  for (int w = 0; w < 4; ++w) {
    int ls = l - 3 + w;
    if (ls >= 0) acc += xz[(rowBase + ls) * (size_t)(2 * DINNER) + d] * cw[d * 4 + w];
  }
  float v = silu_f(acc);
  xc[rl * (size_t)DINNER + d] = v;
  xcb[rl * (size_t)DINNER + d] = (__bf16)v;
}

// ---------------------------------------------------------------------------
// Selective scan: 1 thread per (batch, channel); 16-state recurrence in
// registers; emits gated y directly as bf16 (out_proj GEMM operand).
// ---------------------------------------------------------------------------
__global__ __launch_bounds__(256)
void scan_kernel(const float* __restrict__ xdbl, const float* __restrict__ xz,
                 const float* __restrict__ xc, const float* __restrict__ delta,
                 __bf16* __restrict__ ybf,
                 const float* __restrict__ A_log, const float* __restrict__ Dp) {
  const int t = (int)blockIdx.x * 256 + (int)threadIdx.x;   // 0 .. B*DINNER-1
  if (t >= B_SZ * DINNER) return;
  const int bi = t / DINNER;
  const int d = t % DINNER;

  float Ac[DSTATE], h[DSTATE];
#pragma unroll
  for (int n = 0; n < DSTATE; ++n) {
    Ac[n] = -__expf(A_log[d * DSTATE + n]);
    h[n] = 0.0f;
  }
  const float Dd = Dp[d];

  for (int l = 0; l < LSEQ; ++l) {
    const size_t row = (size_t)bi * LSEQ + l;
    const float dt = delta[row * DINNER + d];
    const float u  = xc[row * DINNER + d];
    const float zv = xz[row * (size_t)(2 * DINNER) + DINNER + d];
    const float4* B4 = (const float4*)(const void*)(xdbl + row * 96 + DTRANK);
    const float4* C4 = B4 + 4;
    float y = 0.0f;
#pragma unroll
    for (int q = 0; q < 4; ++q) {
      float4 Bq = B4[q];
      float4 Cq = C4[q];
      float Bv[4] = {Bq.x, Bq.y, Bq.z, Bq.w};
      float Cv[4] = {Cq.x, Cq.y, Cq.z, Cq.w};
#pragma unroll
      for (int s = 0; s < 4; ++s) {
        int n = q * 4 + s;
        float dA = __expf(dt * Ac[n]);
        h[n] = dA * h[n] + dt * Bv[s] * u;
        y += h[n] * Cv[s];
      }
    }
    y = (y + u * Dd) * silu_f(zv);
    ybf[row * DINNER + d] = (__bf16)y;
  }
}

// ---------------------------------------------------------------------------
// Orchestration
// ---------------------------------------------------------------------------
extern "C" void kernel_launch(void* const* d_in, const int* in_sizes, int n_in,
                              void* d_out, int out_size, void* d_ws, size_t ws_size,
                              hipStream_t stream) {
  (void)in_sizes; (void)n_in; (void)out_size; (void)ws_size;

  const float* x  = (const float*)d_in[0];
  const float* ng = (const float*)d_in[1];
  const float* nb = (const float*)d_in[2];
  float* out = (float*)d_out;

  // Workspace carving (256B-aligned buffers)
  char* wp = (char*)d_ws;
  auto carve = [&](size_t bytes) {
    char* r = wp;
    wp += (bytes + 255) & ~(size_t)255;
    return (void*)r;
  };
  float*  xz      = (float*)carve((size_t)NROWS * 2 * DINNER * 4);
  float*  xc      = (float*)carve((size_t)NROWS * DINNER * 4);
  float*  xdbl    = (float*)carve((size_t)NROWS * 96 * 4);
  float*  delta   = (float*)carve((size_t)NROWS * DINNER * 4);
  __bf16* xn_bf   = (__bf16*)carve((size_t)NROWS * DMODEL * 2);
  __bf16* xnf_bf  = (__bf16*)carve((size_t)NROWS * DMODEL * 2);
  __bf16* xc_bf   = (__bf16*)carve((size_t)NROWS * DINNER * 2);
  __bf16* xdbl_bf = (__bf16*)carve((size_t)NROWS * 96 * 2);
  __bf16* y_bf    = (__bf16*)carve((size_t)NROWS * DINNER * 2);
  __bf16* w_in_bf  = (__bf16*)carve((size_t)2 * DINNER * DMODEL * 2);
  __bf16* w_x_bf   = (__bf16*)carve((size_t)96 * DINNER * 2);
  __bf16* w_dt_bf  = (__bf16*)carve((size_t)DINNER * DTRANK * 2);
  __bf16* w_out_bf = (__bf16*)carve((size_t)DMODEL * DINNER * 2);

  auto blocks = [](size_t n) { return dim3((unsigned)((n + 255) / 256)); };

  // 1) residual init: out = x
  {
    size_t n = (size_t)NROWS * DMODEL;
    copy_kernel<<<blocks(n), dim3(256), 0, stream>>>(x, out, n);
  }
  // 2) layernorm -> bf16 xn / xnf
  ln_kernel<<<dim3(NROWS), dim3(256), 0, stream>>>(x, ng, nb, xn_bf, xnf_bf);

  for (int dir = 0; dir < 2; ++dir) {
    const __bf16* Xin = dir ? xnf_bf : xn_bf;
    const int p = 3 + dir * 9;
    const float* in_proj_w  = (const float*)d_in[p + 0];  // [4096,1024]
    const float* conv_w     = (const float*)d_in[p + 1];  // [2048,4]
    const float* conv_b     = (const float*)d_in[p + 2];  // [2048]
    const float* x_proj_w   = (const float*)d_in[p + 3];  // [96,2048]
    const float* dt_proj_w  = (const float*)d_in[p + 4];  // [2048,64]
    const float* dt_proj_b  = (const float*)d_in[p + 5];  // [2048]
    const float* A_log      = (const float*)d_in[p + 6];  // [2048,16]
    const float* Dparam     = (const float*)d_in[p + 7];  // [2048]
    const float* out_proj_w = (const float*)d_in[p + 8];  // [1024,2048]

    // weight casts to bf16 (reused buffers per direction)
    cast_kernel<<<blocks((size_t)2 * DINNER * DMODEL), dim3(256), 0, stream>>>(
        in_proj_w, w_in_bf, (size_t)2 * DINNER * DMODEL);
    cast_kernel<<<blocks((size_t)96 * DINNER), dim3(256), 0, stream>>>(
        x_proj_w, w_x_bf, (size_t)96 * DINNER);
    cast_kernel<<<blocks((size_t)DINNER * DTRANK), dim3(256), 0, stream>>>(
        dt_proj_w, w_dt_bf, (size_t)DINNER * DTRANK);
    cast_kernel<<<blocks((size_t)DMODEL * DINNER), dim3(256), 0, stream>>>(
        out_proj_w, w_out_bf, (size_t)DMODEL * DINNER);

    // xz = Xin @ in_proj_w^T   [8192 x 4096] f32
    gemm_wmma_bf16<<<dim3(2 * DINNER / BN, NROWS / BM), dim3(256), 0, stream>>>(
        Xin, DMODEL, w_in_bf, DMODEL, xz, 2 * DINNER, nullptr,
        NROWS, 2 * DINNER, DMODEL, 0, nullptr);

    // xc = silu(causal depthwise conv(xi) + b)  (f32 + bf16)
    {
      size_t n = (size_t)NROWS * DINNER;
      conv_silu_kernel<<<blocks(n), dim3(256), 0, stream>>>(xz, conv_w, conv_b, xc, xc_bf);
    }

    // x_dbl = xc @ x_proj_w^T   [8192 x 96] f32 + bf16 shadow
    gemm_wmma_bf16<<<dim3((96 + BN - 1) / BN, NROWS / BM), dim3(256), 0, stream>>>(
        xc_bf, DINNER, w_x_bf, DINNER, xdbl, 96, xdbl_bf,
        NROWS, 96, DINNER, 0, nullptr);

    // delta = softplus(x_dbl[:, :64] @ dt_proj_w^T + dt_proj_b)   [8192 x 2048]
    gemm_wmma_bf16<<<dim3(DINNER / BN, NROWS / BM), dim3(256), 0, stream>>>(
        xdbl_bf, 96, w_dt_bf, DTRANK, delta, DINNER, nullptr,
        NROWS, DINNER, DTRANK, 1, dt_proj_b);

    // selective scan -> gated y (bf16)
    scan_kernel<<<dim3((B_SZ * DINNER) / 256), dim3(256), 0, stream>>>(
        xdbl, xz, xc, delta, y_bf, A_log, Dparam);

    // out (+)= y @ out_proj_w^T, bwd direction un-flips rows on store
    gemm_wmma_bf16<<<dim3(DMODEL / BN, NROWS / BM), dim3(256), 0, stream>>>(
        y_bf, DINNER, w_out_bf, DINNER, out, DMODEL, nullptr,
        NROWS, DMODEL, DINNER, dir ? 3 : 2, nullptr);
  }
}
